// GatedRealLRUBPTTMin_13228499271926
// MI455X (gfx1250) — matrix operand order repaired and checked by reading, not compiled, not run on hardware
//
#include <hip/hip_runtime.h>
#include <hip/hip_bf16.h>

// ---------------------------------------------------------------------------
// Problem constants
// ---------------------------------------------------------------------------
#define T_LEN   4096
#define BATCH   32
#define D_IN    256
#define S_DIM   256
#define IN_DIM  257            // D_IN + 1 (bias/flag column)
#define M_TOT   (T_LEN * BATCH)        // 131072 GEMM rows
#define BS      (BATCH * S_DIM)        // 8192 scan lanes
#define NCHUNK  32
#define CHUNK_L (T_LEN / NCHUNK)       // 128

typedef __attribute__((ext_vector_type(16))) __bf16 v16bf;
typedef __attribute__((ext_vector_type(8)))  float  v8f;

// float -> bf16 (round-to-nearest-even)
__device__ __forceinline__ unsigned f2bf(float f) {
    union { float f; unsigned u; } v; v.f = f;
    unsigned r = v.u + 0x7FFFu + ((v.u >> 16) & 1u);
    return (r >> 16) & 0xFFFFu;
}

// ---------------------------------------------------------------------------
// Kernel 1: repack x [M_TOT, 256] fp32 -> bf16 WMMA A-fragments (16x32 tiles)
// ISA 16-bit A 16x32 layout: lane L -> m = L%16, half h = L/16,
// VGPR i (0..7) holds K pair starting at kbase = (i>=4?16:0) + h*8 + (i&3)*2
// Blob: uint index = ((mt*8 + kt)*32 + lane)*8 + i  (one uint = 2 bf16)
// ---------------------------------------------------------------------------
__global__ void repack_x_kernel(const float* __restrict__ x,
                                unsigned* __restrict__ Apack) {
    unsigned tid = blockIdx.x * blockDim.x + threadIdx.x;   // < 16,777,216
    unsigned i    = tid & 7u;
    unsigned lane = (tid >> 3) & 31u;
    unsigned kt   = (tid >> 8) & 7u;
    unsigned mt   = tid >> 11;
    unsigned h    = lane >> 4;
    unsigned m    = mt * 16u + (lane & 15u);
    unsigned k    = kt * 32u + ((i >= 4u) ? 16u : 0u) + h * 8u + (i & 3u) * 2u;
    const float* p = x + (size_t)m * D_IN + k;
    Apack[tid] = f2bf(p[0]) | (f2bf(p[1]) << 16);
}

// ---------------------------------------------------------------------------
// Kernel 2: repack three weights [S, 257] fp32 -> bf16 WMMA B-fragments
// ISA 16-bit B 32x16 layout: lane L -> n = L%16, half h = L/16,
// VGPR j (0..7) holds K pair starting at kbase = (j>=4?16:0) + h*8 + (j&3)*2
// Blob: uint index = (((w*16 + nt)*8 + kt)*32 + lane)*8 + j
// Also extracts bias column (k = 256) into fp32 bias[3][256].
// w: 0 = V_nu (pre), 1 = B (bx), 2 = gate_W (gx)
// ---------------------------------------------------------------------------
__global__ void repack_w_kernel(const float* __restrict__ Vw,
                                const float* __restrict__ Bw,
                                const float* __restrict__ Gw,
                                unsigned* __restrict__ Wpack,
                                float* __restrict__ bias) {
    unsigned tid = blockIdx.x * blockDim.x + threadIdx.x;
    if (tid < 3u * 16u * 8u * 32u * 8u) {                   // 98304 uints
        unsigned j    = tid & 7u;
        unsigned lane = (tid >> 3) & 31u;
        unsigned kt   = (tid >> 8) & 7u;
        unsigned nt   = (tid >> 11) & 15u;
        unsigned w    = tid >> 15;
        const float* W = (w == 0u) ? Vw : ((w == 1u) ? Bw : Gw);
        unsigned h = lane >> 4;
        unsigned n = nt * 16u + (lane & 15u);
        unsigned k = kt * 32u + ((j >= 4u) ? 16u : 0u) + h * 8u + (j & 3u) * 2u;
        const float* p = W + (size_t)n * IN_DIM + k;
        Wpack[tid] = f2bf(p[0]) | (f2bf(p[1]) << 16);
    }
    if (tid < 3u * S_DIM) {                                  // bias columns
        unsigned w = tid >> 8, s = tid & 255u;
        const float* W = (w == 0u) ? Vw : ((w == 1u) ? Bw : Gw);
        bias[tid] = W[(size_t)s * IN_DIM + D_IN];
    }
}

// ---------------------------------------------------------------------------
// Kernel 3: fused WMMA GEMM (3 projections sharing A) + gating epilogue.
// One wave = one 16x16 (m,n) tile; 8 waves/block cover 8 consecutive m-tiles.
// 8 K-steps x 3 matrices = 24 v_wmma_f32_16x16x32_bf16 per wave.
// Emits lam[T*B, S] and u[T*B, S] in fp32.
// ---------------------------------------------------------------------------
__global__ void __launch_bounds__(256)
gemm_gate_kernel(const __bf16* __restrict__ Apack,
                 const __bf16* __restrict__ Wpack,
                 const float* __restrict__ bias,
                 const float* __restrict__ nu_log,
                 float* __restrict__ lam_ws,
                 float* __restrict__ u_ws) {
    const int lane = threadIdx.x & 31;
    const int wave = threadIdx.x >> 5;                // 0..7
    const int mt   = blockIdx.x * 8 + wave;           // 0..8191
    const int nt   = blockIdx.y;                      // 0..15

    v8f accP = {}, accB = {}, accG = {};

    const __bf16* aBase = Apack + (size_t)mt * (8 * 512) + lane * 16;
    const __bf16* wV = Wpack + ((size_t)(0 * 16 + nt) * 8) * 512 + lane * 16;
    const __bf16* wB = Wpack + ((size_t)(1 * 16 + nt) * 8) * 512 + lane * 16;
    const __bf16* wG = Wpack + ((size_t)(2 * 16 + nt) * 8) * 512 + lane * 16;

#pragma unroll
    for (int kt = 0; kt < 8; ++kt) {
        v16bf a  = *reinterpret_cast<const v16bf*>(aBase + kt * 512);
        v16bf bv = *reinterpret_cast<const v16bf*>(wV + kt * 512);
        v16bf bb = *reinterpret_cast<const v16bf*>(wB + kt * 512);
        v16bf bg = *reinterpret_cast<const v16bf*>(wG + kt * 512);
        accP = __builtin_amdgcn_wmma_f32_16x16x32_bf16(false, a, false, bv,
                                                       (short)0, accP, false, false);
        accB = __builtin_amdgcn_wmma_f32_16x16x32_bf16(false, a, false, bb,
                                                       (short)0, accB, false, false);
        accG = __builtin_amdgcn_wmma_f32_16x16x32_bf16(false, a, false, bg,
                                                       (short)0, accG, false, false);
    }

    // C/D layout: VGPR r -> m = r + 8*(lane>=16), n = lane%16
    const int n = lane & 15;
    const int s = nt * 16 + n;
    const int mBase = mt * 16 + ((lane >= 16) ? 8 : 0);
    const float bV = bias[0 * S_DIM + s];
    const float bB = bias[1 * S_DIM + s];
    const float bG = bias[2 * S_DIM + s];
    const float alpha_nu = 3.0f * __expf(nu_log[s]);

#pragma unroll
    for (int r = 0; r < 8; ++r) {
        const int   m   = mBase + r;
        const float pre = accP[r] + bV;
        const float bx  = accB[r] + bB;
        const float gx  = accG[r] + bG;
        const float sig = 1.0f / (1.0f + __expf(-pre));
        const float l   = __expf(-alpha_nu * (1.0f - sig));
        const float gam = sqrtf(1.0f - l * l + 1e-7f);
        const float uu  = gam * bx * gx;
        const size_t idx = (size_t)m * S_DIM + s;
        lam_ws[idx] = l;
        u_ws[idx]   = uu;
    }
}

// ---------------------------------------------------------------------------
// Chunked linear-recurrence scan: h_t = lam_t*h_{t-1} + u_t.
// Pass 1: per-chunk carry P_c = prod(lam), E_c = scan result with h0=0.
// Pass 2: serial combine across 32 chunks -> Hin[c] per lane, + h_final.
// Pass 3: re-scan each chunk starting from Hin[c], write outputs.
// ---------------------------------------------------------------------------
__global__ void scan_pass1(const float* __restrict__ lam,
                           const float* __restrict__ u,
                           float* __restrict__ P, float* __restrict__ E) {
    const int idx = blockIdx.x * blockDim.x + threadIdx.x;  // < NCHUNK*BS
    const int e = idx & (BS - 1);
    const int c = idx >> 13;
    const size_t base = (size_t)c * CHUNK_L * BS + e;
    float p = 1.0f, h = 0.0f;
    for (int t = 0; t < CHUNK_L; ++t) {
        const size_t o = base + (size_t)t * BS;
        const float l = lam[o];
        h = __fmaf_rn(l, h, u[o]);
        p *= l;
    }
    P[idx] = p;
    E[idx] = h;
}

__global__ void scan_pass2(const float* __restrict__ P,
                           const float* __restrict__ E,
                           float* __restrict__ Hin,
                           float* __restrict__ h_final) {
    const int e = blockIdx.x * blockDim.x + threadIdx.x;    // < BS
    float h = 0.0f;
#pragma unroll
    for (int c = 0; c < NCHUNK; ++c) {
        Hin[c * BS + e] = h;
        h = __fmaf_rn(P[c * BS + e], h, E[c * BS + e]);
    }
    h_final[e] = h;
}

__global__ void scan_pass3(const float* __restrict__ lam,
                           const float* __restrict__ u,
                           const float* __restrict__ Hin,
                           float* __restrict__ out) {
    const int idx = blockIdx.x * blockDim.x + threadIdx.x;  // < NCHUNK*BS
    const int e = idx & (BS - 1);
    const int c = idx >> 13;
    const size_t base = (size_t)c * CHUNK_L * BS + e;
    float h = Hin[c * BS + e];
    for (int t = 0; t < CHUNK_L; ++t) {
        const size_t o = base + (size_t)t * BS;
        __builtin_prefetch(lam + o + (size_t)8 * BS, 0, 1);  // global_prefetch_b8
        __builtin_prefetch(u   + o + (size_t)8 * BS, 0, 1);
        h = __fmaf_rn(lam[o], h, u[o]);
        out[o] = h;
    }
}

// ---------------------------------------------------------------------------
// Launch
// d_in: 0=x[T,B,D] f32, 1=nu_log[S] f32, 2=B[S,257] f32, 3=gate_W[S,257] f32,
//       4=V_nu[S,257] f32.  d_out: outputs[T,B,S] ++ h_final[B,S], f32.
// ---------------------------------------------------------------------------
extern "C" void kernel_launch(void* const* d_in, const int* in_sizes, int n_in,
                              void* d_out, int out_size, void* d_ws, size_t ws_size,
                              hipStream_t stream) {
    (void)in_sizes; (void)n_in; (void)out_size; (void)ws_size;

    const float* x      = (const float*)d_in[0];
    const float* nu_log = (const float*)d_in[1];
    const float* Bw     = (const float*)d_in[2];
    const float* Gw     = (const float*)d_in[3];
    const float* Vw     = (const float*)d_in[4];
    float* out = (float*)d_out;
    float* h_final = out + (size_t)T_LEN * BS;   // outputs, then final state

    // Workspace carve-up (bytes), all offsets 1KB-aligned.
    char* ws = (char*)d_ws;
    const size_t LAM_BYTES = (size_t)T_LEN * BS * sizeof(float);     // 128 MB
    float*   lam   = (float*)(ws);
    float*   u     = (float*)(ws + LAM_BYTES);
    __bf16*  Apack = (__bf16*)(ws + 2 * LAM_BYTES);                  // 64 MB
    const size_t APACK_BYTES = (size_t)M_TOT * D_IN * sizeof(unsigned short);
    __bf16*  Wpack = (__bf16*)(ws + 2 * LAM_BYTES + APACK_BYTES);    // 384 KB
    const size_t WPACK_BYTES = (size_t)3 * S_DIM * D_IN * sizeof(unsigned short);
    float*   bias  = (float*)(ws + 2 * LAM_BYTES + APACK_BYTES + WPACK_BYTES);
    const size_t BIAS_BYTES = 1024;                                  // 3 KB used, pad
    char* tail = ws + 2 * LAM_BYTES + APACK_BYTES + WPACK_BYTES + 4 * BIAS_BYTES;
    float* P   = (float*)(tail);
    float* E   = (float*)(tail + (size_t)NCHUNK * BS * sizeof(float));
    float* Hin = (float*)(tail + (size_t)2 * NCHUNK * BS * sizeof(float));

    // 1) repack x -> bf16 A fragments (16,777,216 uint writes)
    repack_x_kernel<<<(M_TOT / 16) * 8 * 32 * 8 / 256, 256, 0, stream>>>(
        x, (unsigned*)Apack);

    // 2) repack weights -> bf16 B fragments + bias
    repack_w_kernel<<<(98304 + 255) / 256, 256, 0, stream>>>(
        Vw, Bw, Gw, (unsigned*)Wpack, bias);

    // 3) fused WMMA GEMM + gating epilogue
    dim3 ggrid(M_TOT / 16 / 8, S_DIM / 16);   // (1024, 16)
    gemm_gate_kernel<<<ggrid, 256, 0, stream>>>(Apack, Wpack, bias, nu_log,
                                                lam, u);

    // 4) chunked scan
    scan_pass1<<<(NCHUNK * BS) / 256, 256, 0, stream>>>(lam, u, P, E);
    scan_pass2<<<BS / 256, 256, 0, stream>>>(P, E, Hin, h_final);
    scan_pass3<<<(NCHUNK * BS) / 256, 256, 0, stream>>>(lam, u, Hin, out);
}
